// TemporalPhaseManifold_68891275427896
// MI455X (gfx1250) — compile-verified, matrix operand order checked by reading, not run on hardware
//
#include <hip/hip_runtime.h>
#include <hip/hip_bf16.h>

// ---------------------------------------------------------------------------
// TemporalPhaseManifold forward for gfx1250 (MI455X), wave32 + WMMA bf16.
//
// Shapes: B=4 S=1024 D_IN=256 M=1024 H=16 HD=64 F=8 L=2 D_OUT=256, T=B*S=4096
//
// Exact algebraic simplifications:
//  * mean_F(softmax(..)) == 1/F, so pf = |x@Wemb + bemb| * 0.125 (Wamp unused)
//  * cos(pc_s - pc_t) bias is rank-16 -> folded into QK^T by augmenting the
//    head dim 64 -> 96 (64 + 8cos + 8sin + 16 zeros so K % 32 == 0).
//
// GEMMs: 128x128 block tile, 4 waves (64x64 each), BK=32, double-buffered LDS
// with register staging; epilogues fully compile-time specialized.
// Attention: flash-style streaming softmax, never materializes [B,H,S,S].
// ---------------------------------------------------------------------------

typedef __bf16 bf16;
typedef __attribute__((ext_vector_type(16))) __bf16 v16bf;
typedef __attribute__((ext_vector_type(8)))  float  v8f;

struct alignas(16) Q4 { unsigned x, y, z, w; };
union Frag { Q4 q[2]; v16bf v; };

__device__ __forceinline__ v16bf frag_ld(const bf16* t, int row0, int stride, int koff) {
    // 16-bit A/B fragment: lane&15 selects row(A)/col(B); lane>>4 selects K half.
    // Per lane: halves [koff+kb .. +7] then [koff+kb+16 .. +23], kb = 8*(lane>>4).
    const int lane = threadIdx.x & 31;
    const bf16* p = t + (row0 + (lane & 15)) * stride + koff + ((lane >> 4) << 3);
    Frag f;
    f.q[0] = *(const Q4*)p;
    f.q[1] = *(const Q4*)(p + 16);
    return f.v;
}

__device__ __forceinline__ v8f wmma_bf(v16bf a, v16bf b, v8f c) {
    return __builtin_amdgcn_wmma_f32_16x16x32_bf16(false, a, false, b, (short)0, c,
                                                   false, false);
}

// ---------------------------------------------------------------------------
// bf16 WMMA GEMM: C[Mr,N] = A[Mr,K] @ BT[N,K]^T + bias[N] (+res) [+abs/relu]
// MODE: 0 none, 1 |val|*0.125, 2 relu. Mr,N % 128 == 0, K % 32 == 0.
// ---------------------------------------------------------------------------
template <int MODE, bool HAS_RES, bool HAS_CF, bool HAS_CB>
__global__ __launch_bounds__(128) void k_gemm(
    const bf16* __restrict__ A, const bf16* __restrict__ BT,
    const float* __restrict__ bias, const float* __restrict__ res,
    float* __restrict__ Cf, bf16* __restrict__ Cb, int Mr, int N, int K)
{
    __shared__ __attribute__((aligned(16))) bf16 sA[2][128 * 32];
    __shared__ __attribute__((aligned(16))) bf16 sB[2][128 * 32];

    const int tid = threadIdx.x;
    const int m0 = blockIdx.y * 128, n0 = blockIdx.x * 128;
    const int wid = tid >> 5;
    const int wm = (wid >> 1) * 64, wn = (wid & 1) * 64;

    v8f acc[4][4];
#pragma unroll
    for (int i = 0; i < 4; i++)
#pragma unroll
        for (int j = 0; j < 4; j++)
#pragma unroll
            for (int e = 0; e < 8; e++) acc[i][j][e] = 0.f;

    // per-thread tile-chunk coordinates (4 x b128 per operand per stage)
    Q4 ra[4], rb[4];
    int rr[4], kk4[4];
#pragma unroll
    for (int i = 0; i < 4; i++) {
        int c = i * 128 + tid;            // 512 chunks of 8 halves = 128x32
        rr[i] = c >> 2;
        kk4[i] = (c & 3) << 3;
    }

    // prologue: stage 0 -> LDS buf 0
#pragma unroll
    for (int i = 0; i < 4; i++) {
        ra[i] = *(const Q4*)&A[(size_t)(m0 + rr[i]) * K + kk4[i]];
        rb[i] = *(const Q4*)&BT[(size_t)(n0 + rr[i]) * K + kk4[i]];
    }
#pragma unroll
    for (int i = 0; i < 4; i++) {
        *(Q4*)&sA[0][rr[i] * 32 + kk4[i]] = ra[i];
        *(Q4*)&sB[0][rr[i] * 32 + kk4[i]] = rb[i];
    }
    __syncthreads();

    int cur = 0;
    for (int k0 = 0; k0 < K; k0 += 32) {
        const bool has_next = (k0 + 32) < K;
        if (has_next) {                   // global loads fly during the WMMAs
#pragma unroll
            for (int i = 0; i < 4; i++) {
                ra[i] = *(const Q4*)&A[(size_t)(m0 + rr[i]) * K + k0 + 32 + kk4[i]];
                rb[i] = *(const Q4*)&BT[(size_t)(n0 + rr[i]) * K + k0 + 32 + kk4[i]];
            }
        }

        v16bf af[4], bfr[4];
#pragma unroll
        for (int i = 0; i < 4; i++) af[i]  = frag_ld(&sA[cur][0], wm + i * 16, 32, 0);
#pragma unroll
        for (int j = 0; j < 4; j++) bfr[j] = frag_ld(&sB[cur][0], wn + j * 16, 32, 0);
#pragma unroll
        for (int i = 0; i < 4; i++)
#pragma unroll
            for (int j = 0; j < 4; j++)
                acc[i][j] = wmma_bf(af[i], bfr[j], acc[i][j]);

        if (has_next) {
#pragma unroll
            for (int i = 0; i < 4; i++) {
                *(Q4*)&sA[cur ^ 1][rr[i] * 32 + kk4[i]] = ra[i];
                *(Q4*)&sB[cur ^ 1][rr[i] * 32 + kk4[i]] = rb[i];
            }
        }
        __syncthreads();
        cur ^= 1;
    }

    // epilogue (branch-free; predicates are template constants)
    const int lane = tid & 31;
    const int rsel = (lane >> 4) << 3;    // C layout: row = v + 8*(lane>=16)
    const int csel = lane & 15;           //           col = lane&15
    float bcol[4];
#pragma unroll
    for (int j = 0; j < 4; j++) bcol[j] = bias[n0 + wn + j * 16 + csel];

#pragma unroll
    for (int i = 0; i < 4; i++)
#pragma unroll
        for (int j = 0; j < 4; j++)
#pragma unroll
            for (int v = 0; v < 8; v++) {
                int row = m0 + wm + i * 16 + v + rsel;
                int col = n0 + wn + j * 16 + csel;
                size_t o = (size_t)row * N + col;
                float val = acc[i][j][v] + bcol[j];
                if constexpr (HAS_RES) val += res[o];
                if constexpr (MODE == 1) val = fabsf(val) * 0.125f;
                if constexpr (MODE == 2) val = fmaxf(val, 0.f);
                if constexpr (HAS_CF) Cf[o] = val;
                if constexpr (HAS_CB) Cb[o] = (bf16)val;
            }
}

// ---------------------------------------------------------------------------
// Flash attention, augmented head dim 96 (=3*32). Block: 128 thr (4 waves),
// 64 query rows per block, streams K/V in 64-row tiles. grid=(S/64, B*H).
// Qa/Ka: [B*H, S, 96] bf16. Vb: [B*H, S, 64] bf16. Ob: [B*S, H*64] bf16.
// ---------------------------------------------------------------------------
__global__ __launch_bounds__(128) void k_attn(
    const bf16* __restrict__ Qa, const bf16* __restrict__ Ka,
    const bf16* __restrict__ Vb, bf16* __restrict__ Ob, int S, int H)
{
    __shared__ __attribute__((aligned(16))) bf16 sQ[64 * 96];
    __shared__ __attribute__((aligned(16))) bf16 sK[64 * 96];
    __shared__ __attribute__((aligned(16))) bf16 sVT[64 * 64];   // [d][t]
    __shared__ __attribute__((aligned(16))) bf16 sP[4][16 * 64]; // per-wave P

    const int tid = threadIdx.x, lane = tid & 31, wid = tid >> 5;
    const int bh = blockIdx.y;
    const int b = bh / H, h = bh % H;
    const int q0 = blockIdx.x * 64;

    const size_t baseQ = ((size_t)bh * S + q0) * 96;
#pragma unroll
    for (int i = 0; i < 6; i++) {                       // 64x96 halves
        int c = i * 128 + tid;
        int r = c / 12, kc = (c % 12) * 8;
        *(Q4*)&sQ[r * 96 + kc] = *(const Q4*)&Qa[baseQ + (size_t)r * 96 + kc];
    }

    float mi[8], li[8];
    v8f oacc[4];
#pragma unroll
    for (int v = 0; v < 8; v++) { mi[v] = -1e30f; li[v] = 0.f; }
#pragma unroll
    for (int j = 0; j < 4; j++)
#pragma unroll
        for (int e = 0; e < 8; e++) oacc[j][e] = 0.f;

    for (int t0 = 0; t0 < S; t0 += 64) {
        const size_t baseK = ((size_t)bh * S + t0) * 96;
#pragma unroll
        for (int i = 0; i < 6; i++) {
            int c = i * 128 + tid;
            int r = c / 12, kc = (c % 12) * 8;
            *(Q4*)&sK[r * 96 + kc] = *(const Q4*)&Ka[baseK + (size_t)r * 96 + kc];
        }
        const size_t baseV = ((size_t)bh * S + t0) * 64;
#pragma unroll
        for (int i = 0; i < 4; i++) {                   // transpose V into LDS
            int c = i * 128 + tid;
            int r = c >> 3, d0 = (c & 7) << 3;
            Q4 qv = *(const Q4*)&Vb[baseV + (size_t)r * 64 + d0];
            const bf16* pv = (const bf16*)&qv;
#pragma unroll
            for (int e = 0; e < 8; e++) sVT[(d0 + e) * 64 + r] = pv[e];
        }
        __syncthreads();

        // scores: 16(q) x 64(t) per wave, K=96 in 3 steps
        v8f sacc[4];
#pragma unroll
        for (int j = 0; j < 4; j++)
#pragma unroll
            for (int e = 0; e < 8; e++) sacc[j][e] = 0.f;
#pragma unroll
        for (int kk = 0; kk < 3; kk++) {
            v16bf aq = frag_ld(sQ, wid * 16, 96, kk * 32);
#pragma unroll
            for (int j = 0; j < 4; j++) {
                v16bf bk = frag_ld(sK, j * 16, 96, kk * 32);
                sacc[j] = wmma_bf(aq, bk, sacc[j]);
            }
        }

        // online softmax; row v+8*(lane>>4); reduce within 16-lane half
#pragma unroll
        for (int v = 0; v < 8; v++) {
            float mx = -1e30f;
#pragma unroll
            for (int j = 0; j < 4; j++) mx = fmaxf(mx, sacc[j][v]);
            mx *= 0.125f;
            for (int off = 1; off < 16; off <<= 1)
                mx = fmaxf(mx, __shfl_xor(mx, off, 32));
            float mn = fmaxf(mi[v], mx);
            float corr = __expf(mi[v] - mn);
            float rs = 0.f;
#pragma unroll
            for (int j = 0; j < 4; j++) {
                float p = __expf(sacc[j][v] * 0.125f - mn);
                sP[wid][(v + ((lane >> 4) << 3)) * 64 + j * 16 + (lane & 15)] = (bf16)p;
                rs += p;
            }
            for (int off = 1; off < 16; off <<= 1)
                rs += __shfl_xor(rs, off, 32);
            li[v] = li[v] * corr + rs;
            mi[v] = mn;
#pragma unroll
            for (int j = 0; j < 4; j++) oacc[j][v] = oacc[j][v] * corr;
        }
        __syncthreads();

        // O += P @ V  (P 16x64 from LDS, V^T tiles)
#pragma unroll
        for (int kk = 0; kk < 2; kk++) {
            v16bf ap = frag_ld(sP[wid], 0, 64, kk * 32);
#pragma unroll
            for (int j = 0; j < 4; j++) {
                v16bf bv = frag_ld(sVT, j * 16, 64, kk * 32);
                oacc[j] = wmma_bf(ap, bv, oacc[j]);
            }
        }
        __syncthreads();
    }

    const int rsel = (lane >> 4) << 3, csel = lane & 15;
#pragma unroll
    for (int j = 0; j < 4; j++)
#pragma unroll
        for (int v = 0; v < 8; v++) {
            int srow = q0 + wid * 16 + v + rsel;
            int col = h * 64 + j * 16 + csel;
            float o = oacc[j][v] / li[v];
            Ob[((size_t)b * S + srow) * (size_t)(H * 64) + col] = (bf16)o;
        }
}

// --------------------------- small helper kernels ---------------------------

__global__ void k_cvt(const float* __restrict__ in, bf16* __restrict__ out, int n) {
    int i = blockIdx.x * 256 + threadIdx.x;
    if (i < n) out[i] = (bf16)in[i];
}

// in [K,N] f32 -> out [N,K] bf16 (transpose + convert)
__global__ void k_cvtT(const float* __restrict__ in, bf16* __restrict__ out, int K, int N) {
    int i = blockIdx.x * 256 + threadIdx.x;
    if (i >= K * N) return;
    int n = i / K, k = i % K;
    out[i] = (bf16)in[(size_t)k * N + n];
}

__global__ void k_tables(const float* __restrict__ freqs, const float* __restrict__ phase_off,
                         float* __restrict__ cs, float* __restrict__ sn, int S) {
    int i = blockIdx.x * 256 + threadIdx.x;
    if (i >= S * 8) return;
    int s = i >> 3, f = i & 7;
    float pc = ((float)s / (float)S) * 6.28318530717958647f * freqs[f] + phase_off[f];
    cs[i] = cosf(pc);
    sn[i] = sinf(pc);
}

// Q/K/V f32 [B*S,1024] -> Qa/Ka [B*H,S,96] bf16 (with cos/sin bias features),
// Vb [B*H,S,64] bf16. grid = B*S blocks.
__global__ __launch_bounds__(128) void k_augment(
    const float* __restrict__ Qf, const float* __restrict__ Kf, const float* __restrict__ Vf,
    const float* __restrict__ cs, const float* __restrict__ sn,
    const float* __restrict__ pbias_l,
    bf16* __restrict__ Qa, bf16* __restrict__ Ka, bf16* __restrict__ Vb, int S, int H)
{
    int row = blockIdx.x;                 // b*S + s
    int b = row / S, s = row % S;
    int tid = threadIdx.x;
    for (int idx = tid; idx < H * 64; idx += 128) {
        int h = idx >> 6, d = idx & 63;
        size_t dst = (size_t)(b * H + h) * S + s;
        float q = Qf[(size_t)row * 1024 + idx];
        float k = Kf[(size_t)row * 1024 + idx];
        float v = Vf[(size_t)row * 1024 + idx];
        Qa[dst * 96 + d] = (bf16)q;
        Ka[dst * 96 + d] = (bf16)k;
        Vb[dst * 64 + d] = (bf16)v;
    }
    if (tid < H) {
        int h = tid;
        size_t dst = (size_t)(b * H + h) * S + s;
#pragma unroll
        for (int f = 0; f < 8; f++) {
            float c = cs[s * 8 + f], sv = sn[s * 8 + f], pb = pbias_l[h * 8 + f];
            Qa[dst * 96 + 64 + f] = (bf16)(8.f * pb * c);   // 1/scale = 8
            Qa[dst * 96 + 72 + f] = (bf16)(8.f * pb * sv);
            Ka[dst * 96 + 64 + f] = (bf16)c;
            Ka[dst * 96 + 72 + f] = (bf16)sv;
        }
#pragma unroll
        for (int f = 80; f < 96; f++) {
            Qa[dst * 96 + f] = (bf16)0.f;
            Ka[dst * 96 + f] = (bf16)0.f;
        }
    }
}

// LayerNorm over rows of y [T,1024]; writes pf f32 + bf16.
__global__ __launch_bounds__(256) void k_lnorm(
    const float* __restrict__ y, const float* __restrict__ g, const float* __restrict__ bta,
    float* __restrict__ pf, bf16* __restrict__ pfb, int N)
{
    __shared__ float s1[8], s2[8];
    int row = blockIdx.x, tid = threadIdx.x, lane = tid & 31, wid = tid >> 5;
    const float* yr = y + (size_t)row * N;
    float a = 0.f, q = 0.f;
    for (int i = tid; i < N; i += 256) { float v = yr[i]; a += v; q += v * v; }
    for (int off = 1; off < 32; off <<= 1) {
        a += __shfl_xor(a, off, 32);
        q += __shfl_xor(q, off, 32);
    }
    if (lane == 0) { s1[wid] = a; s2[wid] = q; }
    __syncthreads();
    if (tid == 0) {
        float ta = 0.f, tq = 0.f;
        for (int i = 0; i < 8; i++) { ta += s1[i]; tq += s2[i]; }
        float mu = ta / N;
        float var = tq / N - mu * mu;
        s1[0] = mu;
        s2[0] = rsqrtf(var + 1e-5f);
    }
    __syncthreads();
    float mu = s1[0], inv = s2[0];
    for (int i = tid; i < N; i += 256) {
        float v = (yr[i] - mu) * inv * g[i] + bta[i];
        pf[(size_t)row * N + i] = v;
        pfb[(size_t)row * N + i] = (bf16)v;
    }
}

// ---------------------------------------------------------------------------

extern "C" void kernel_launch(void* const* d_in, const int* in_sizes, int n_in,
                              void* d_out, int out_size, void* d_ws, size_t ws_size,
                              hipStream_t stream) {
    constexpr int B = 4, S = 1024, DIN = 256, M = 1024, H = 16, L = 2, DOUT = 256;
    constexpr int T = B * S;               // 4096 tokens
    constexpr int MH = M / 2;              // 512 MLP hidden

    const float* x     = (const float*)d_in[0];
    const float* Wemb  = (const float*)d_in[1];
    const float* bemb  = (const float*)d_in[2];
    // d_in[3]=Wamp, d_in[4]=bamp: mathematically unused (mean of softmax = 1/F)
    const float* freqs = (const float*)d_in[5];
    const float* phoff = (const float*)d_in[6];
    const float* Wq    = (const float*)d_in[7];
    const float* bq    = (const float*)d_in[8];
    const float* Wk    = (const float*)d_in[9];
    const float* bk    = (const float*)d_in[10];
    const float* Wv    = (const float*)d_in[11];
    const float* bv    = (const float*)d_in[12];
    const float* Wo    = (const float*)d_in[13];
    const float* bo    = (const float*)d_in[14];
    const float* pbias = (const float*)d_in[15];
    const float* ln_g  = (const float*)d_in[16];
    const float* ln_b  = (const float*)d_in[17];
    const float* W1    = (const float*)d_in[18];
    const float* b1    = (const float*)d_in[19];
    const float* W2    = (const float*)d_in[20];
    const float* b2    = (const float*)d_in[21];
    float* out = (float*)d_out;

    // workspace bump allocator (256B aligned)
    char* w = (char*)d_ws;
    auto alloc = [&](size_t bytes) -> void* {
        void* p = (void*)w;
        w += (bytes + 255) & ~(size_t)255;
        return p;
    };
    bf16* xb    = (bf16*)alloc((size_t)T * DIN * 2);
    bf16* WembT = (bf16*)alloc((size_t)M * DIN * 2);
    bf16* WqT[2], *WkT[2], *WvT[2], *WoT[2];
    for (int l = 0; l < L; l++) {
        WqT[l] = (bf16*)alloc((size_t)M * M * 2);
        WkT[l] = (bf16*)alloc((size_t)M * M * 2);
        WvT[l] = (bf16*)alloc((size_t)M * M * 2);
        WoT[l] = (bf16*)alloc((size_t)M * M * 2);
    }
    bf16* W1T = (bf16*)alloc((size_t)MH * M * 2);
    bf16* W2T = (bf16*)alloc((size_t)DOUT * MH * 2);
    float* pf_f = (float*)alloc((size_t)T * M * 4);
    bf16*  pf_b = (bf16*)alloc((size_t)T * M * 2);
    float* csb = (float*)alloc((size_t)S * 8 * 4);
    float* snb = (float*)alloc((size_t)S * 8 * 4);
    float* Qf = (float*)alloc((size_t)T * M * 4);
    float* Kf = (float*)alloc((size_t)T * M * 4);
    float* Vf = (float*)alloc((size_t)T * M * 4);
    bf16* Qa = (bf16*)alloc((size_t)B * H * S * 96 * 2);
    bf16* Ka = (bf16*)alloc((size_t)B * H * S * 96 * 2);
    bf16* Vb = (bf16*)alloc((size_t)B * H * S * 64 * 2);
    bf16* attb = (bf16*)alloc((size_t)T * M * 2);
    bf16* hdnb = (bf16*)alloc((size_t)T * MH * 2);
    float* yf = Qf;   // reuse Qf for post-attention residual sum

    // --- precompute: bf16 conversions + phase tables ---
    k_cvt<<<(T * DIN + 255) / 256, 256, 0, stream>>>(x, xb, T * DIN);
    k_cvtT<<<(DIN * M + 255) / 256, 256, 0, stream>>>(Wemb, WembT, DIN, M);
    for (int l = 0; l < L; l++) {
        k_cvtT<<<(M * M + 255) / 256, 256, 0, stream>>>(Wq + (size_t)l * M * M, WqT[l], M, M);
        k_cvtT<<<(M * M + 255) / 256, 256, 0, stream>>>(Wk + (size_t)l * M * M, WkT[l], M, M);
        k_cvtT<<<(M * M + 255) / 256, 256, 0, stream>>>(Wv + (size_t)l * M * M, WvT[l], M, M);
        k_cvtT<<<(M * M + 255) / 256, 256, 0, stream>>>(Wo + (size_t)l * M * M, WoT[l], M, M);
    }
    k_cvtT<<<(M * MH + 255) / 256, 256, 0, stream>>>(W1, W1T, M, MH);
    k_cvtT<<<(MH * DOUT + 255) / 256, 256, 0, stream>>>(W2, W2T, MH, DOUT);
    k_tables<<<(S * 8 + 255) / 256, 256, 0, stream>>>(freqs, phoff, csb, snb, S);

    // --- embedding + phase collapse: pf = |x@Wemb + bemb| * 0.125 ---
    {
        dim3 g(M / 128, T / 128);
        k_gemm<1, false, true, true><<<g, 128, 0, stream>>>(
            xb, WembT, bemb, nullptr, pf_f, pf_b, T, M, DIN);
    }

    // --- transformer layers ---
    for (int l = 0; l < L; l++) {
        dim3 g(M / 128, T / 128);
        k_gemm<0, false, true, false><<<g, 128, 0, stream>>>(
            pf_b, WqT[l], bq + l * M, nullptr, Qf, nullptr, T, M, M);
        k_gemm<0, false, true, false><<<g, 128, 0, stream>>>(
            pf_b, WkT[l], bk + l * M, nullptr, Kf, nullptr, T, M, M);
        k_gemm<0, false, true, false><<<g, 128, 0, stream>>>(
            pf_b, WvT[l], bv + l * M, nullptr, Vf, nullptr, T, M, M);
        k_augment<<<T, 128, 0, stream>>>(Qf, Kf, Vf, csb, snb, pbias + l * H * 8,
                                         Qa, Ka, Vb, S, H);
        {
            dim3 ga(S / 64, B * H);
            k_attn<<<ga, 128, 0, stream>>>(Qa, Ka, Vb, attb, S, H);
        }
        // att @ Wo + bo + residual(pf)
        k_gemm<0, true, true, false><<<g, 128, 0, stream>>>(
            attb, WoT[l], bo + l * M, pf_f, yf, nullptr, T, M, M);
        k_lnorm<<<T, 256, 0, stream>>>(yf, ln_g + l * M, ln_b + l * M, pf_f, pf_b, M);
    }

    // --- output MLP ---
    {
        dim3 g1(MH / 128, T / 128);
        k_gemm<2, false, false, true><<<g1, 128, 0, stream>>>(
            pf_b, W1T, b1, nullptr, nullptr, hdnb, T, MH, M);
        dim3 g2(DOUT / 128, T / 128);
        k_gemm<0, false, true, false><<<g2, 128, 0, stream>>>(
            hdnb, W2T, b2, nullptr, out, nullptr, T, DOUT, MH);
    }

    (void)in_sizes; (void)n_in; (void)out_size; (void)ws_size;
}
